// ODEModel_45603962749807
// MI455X (gfx1250) — matrix operand order repaired and checked by reading, not compiled, not run on hardware
//
#include <hip/hip_runtime.h>

// ---------------------------------------------------------------------------
// MI455X (gfx1250) implementation of the HR-ODE model.
//  - k_fat, k_params : tiny scalar MLPs (negligible work)
//  - k_mlp           : the 100-GFLOP per-timestep MLPs via v_wmma_f32_16x16x32_f16
//  - k_scan1/B/2     : chunked parallel scan of the affine Euler recurrence
// ---------------------------------------------------------------------------

typedef _Float16 f16_t;
typedef __attribute__((ext_vector_type(16))) _Float16 v16h;
typedef __attribute__((ext_vector_type(8)))  _Float16 v8h;
typedef __attribute__((ext_vector_type(4)))  float    v4f;
typedef __attribute__((ext_vector_type(8)))  float    v8f;

#define B_SZ   512
#define T_SZ   3600
#define NTILES ((B_SZ * T_SZ) / 16)   // 115200, tiles never cross a b boundary
#define NW     4                       // waves per block in k_mlp
#define NFRAG  52                      // packed weight fragments in LDS
#define NC     36                      // scan chunks per batch row
#define CL     100                     // steps per chunk (NC*CL == T)

// branchless, numerically stable softplus: max(x,0) + log1p(exp(-|x|))
__device__ __forceinline__ float spf(float x) {
    float e = __expf(-fabsf(x));
    return fmaxf(x, 0.f) + __logf(1.f + e);
}
__device__ __forceinline__ float sigm(float x) {
    return 1.f / (1.f + __expf(-x));
}

// ---- pack a weight matrix W[K][N] into WMMA B-fragment layout in LDS ------
// fragment id = base + nt*nKc + kc ; per lane: col = nt*16 + (lane&15),
// k = kc*32 + (e&7) + 8*(lane>=16) + 16*(e>=8)  (16-bit B layout, 32x16 per fragment)
__device__ __forceinline__ void pack_wfrags(_Float16* ldsW, const float* W,
                                            int base, int Kdim, int Ndim,
                                            int nKc, int nNt, int tid, int nthr) {
    int total = nKc * nNt * 512;             // 32 lanes * 16 elems per frag
    for (int i = tid; i < total; i += nthr) {
        int f    = i >> 9;
        int lane = (i >> 4) & 31;
        int e    = i & 15;
        int kc   = f % nKc;
        int nt   = f / nKc;
        int k    = kc * 32 + (e & 7) + ((lane & 16) >> 1) + ((e & 8) << 1);
        int n    = nt * 16 + (lane & 15);
        float v  = (k < Kdim && n < Ndim) ? W[k * Ndim + n] : 0.f;
        ldsW[((base + f) * 32 + lane) * 16 + e] = (_Float16)v;
    }
}

// ---- A-fragment loaders (16-bit A 16x32 layout) ---------------------------
// lane(0..15): elems 0..7 -> K 0..7, elems 8..15 -> K 16..23
// lane(16..31): +8 on K. row M = lane&15.
// Reload from the per-wave LDS chunk as two aligned 16-byte DS loads.
__device__ __forceinline__ v16h load_afrag_lds(const _Float16* chunkRow, int hi) {
    v8h lo  = *(const v8h*)&chunkRow[hi ? 8 : 0];    // K = hi*8 .. hi*8+7
    v8h hi8 = *(const v8h*)&chunkRow[hi ? 24 : 16];  // K = 16+hi*8 .. 23+hi*8
    return __builtin_shufflevector(lo, hi8, 0, 1, 2, 3, 4, 5, 6, 7,
                                   8, 9, 10, 11, 12, 13, 14, 15);
}

// Layer-1 A fragment from intensity [.,4]: one b128 load, mask with cndmask.
// hi lanes need K=8..11 >= 4 -> all zero; lo lanes need K(e)=e for e<4.
__device__ __forceinline__ v16h load_a1_intensity(const float* src, int row, int hi) {
    v4f r4 = *(const v4f*)&src[(size_t)row * 4];      // 16B-aligned
    v16h a;
#pragma unroll
    for (int e = 0; e < 16; ++e) {
        float v = (e < 4) ? (hi ? 0.f : r4[e & 3]) : 0.f;
        a[e] = (_Float16)v;
    }
    return a;
}

// Layer-1 A fragment from weather [.,3]: three unconditional scalar loads.
__device__ __forceinline__ v16h load_a1_weather(const float* src, int row, int hi) {
    size_t base = (size_t)row * 3;
    float r0 = src[base], r1 = src[base + 1], r2 = src[base + 2];
    v16h a;
#pragma unroll
    for (int e = 0; e < 16; ++e) {
        float v = 0.f;
        if (e == 0) v = hi ? 0.f : r0;
        if (e == 1) v = hi ? 0.f : r1;
        if (e == 2) v = hi ? 0.f : r2;
        a[e] = (_Float16)v;
    }
    return a;
}

// ---- one 3-layer MLP tile: [16 x K1] -> 128 -> 64 -> (<=16), via WMMA -----
__device__ __forceinline__ v8f run_mlp(v16h a1, const _Float16* ldsW,
                                       _Float16 (*chunk)[32],
                                       int base1, int base2, int base3,
                                       const float* __restrict__ b1,
                                       const float* __restrict__ b2,
                                       int lr, int hi, int lane) {
    v16h a2[4];
#pragma unroll
    for (int kc = 0; kc < 4; ++kc) {
        // layer 1 output columns kc*32 .. kc*32+31 (tiles nt=2kc, 2kc+1)
#pragma unroll
        for (int s = 0; s < 2; ++s) {
            int  nt = 2 * kc + s;
            v16h bf = *(const v16h*)&ldsW[((base1 + nt) * 32 + lane) * 16];
            v8f  c  = {};
            c = __builtin_amdgcn_wmma_f32_16x16x32_f16(false, a1, false, bf,
                                                       (short)0, c, false, false);
            float bias = b1[nt * 16 + lr];
#pragma unroll
            for (int r = 0; r < 8; ++r)
                chunk[r + 8 * hi][s * 16 + lr] = (_Float16)spf(c[r] + bias);
        }
        a2[kc] = load_afrag_lds(&chunk[lr][0], hi);   // A-frag for layer-2 K chunk
    }
    v8f c3 = {};
#pragma unroll
    for (int kc3 = 0; kc3 < 2; ++kc3) {
        // layer 2 output columns kc3*32 .. kc3*32+31
#pragma unroll
        for (int s = 0; s < 2; ++s) {
            int nt2 = 2 * kc3 + s;
            v8f c2  = {};
#pragma unroll
            for (int kc = 0; kc < 4; ++kc) {
                v16h bf = *(const v16h*)&ldsW[((base2 + nt2 * 4 + kc) * 32 + lane) * 16];
                c2 = __builtin_amdgcn_wmma_f32_16x16x32_f16(false, a2[kc], false, bf,
                                                            (short)0, c2, false, false);
            }
            float bias = b2[nt2 * 16 + lr];
#pragma unroll
            for (int r = 0; r < 8; ++r)
                chunk[r + 8 * hi][s * 16 + lr] = (_Float16)spf(c2[r] + bias);
        }
        v16h a3 = load_afrag_lds(&chunk[lr][0], hi);
        v16h bf = *(const v16h*)&ldsW[((base3 + kc3) * 32 + lane) * 16];
        c3 = __builtin_amdgcn_wmma_f32_16x16x32_f16(false, a3, false, bf,
                                                    (short)0, c3, false, false);
    }
    return c3;   // bias of layer 3 added by caller
}

// ---------------------------------------------------------------------------
// Heavy kernel: heads/wx MLPs + target = fI*(1 + Bc*fat) + A*wx
// ---------------------------------------------------------------------------
__global__ __launch_bounds__(128) void k_mlp(
    const float* __restrict__ intensity, const float* __restrict__ weather,
    const float* __restrict__ Wa1, const float* __restrict__ ba1,
    const float* __restrict__ Wa2, const float* __restrict__ ba2,
    const float* __restrict__ Wa3, const float* __restrict__ ba3,
    const float* __restrict__ Ww1, const float* __restrict__ bw1,
    const float* __restrict__ Ww2, const float* __restrict__ bw2,
    const float* __restrict__ Ww3, const float* __restrict__ bw3,
    const float* __restrict__ pers, const float* __restrict__ fat,
    const float* __restrict__ prmA, const float* __restrict__ prmB,
    float* __restrict__ target, int ntiles) {

    __shared__ __attribute__((aligned(32))) _Float16 ldsW[NFRAG * 32 * 16]; // 52 KB
    __shared__ __attribute__((aligned(16))) _Float16 schunk[NW][16][32];    // 4 KB
    __shared__ float sC[NW][16][16];                                        // 4 KB

    const int tid = threadIdx.x;
    // fragment bases: a1=0(8) a2=8(16) a3=24(2) w1=26(8) w2=34(16) w3=50(2)
    pack_wfrags(ldsW, Wa1,  0,   4, 128, 1, 8, tid, 128);
    pack_wfrags(ldsW, Wa2,  8, 128,  64, 4, 4, tid, 128);
    pack_wfrags(ldsW, Wa3, 24,  64,   8, 2, 1, tid, 128);
    pack_wfrags(ldsW, Ww1, 26,   3, 128, 1, 8, tid, 128);
    pack_wfrags(ldsW, Ww2, 34, 128,  64, 4, 4, tid, 128);
    pack_wfrags(ldsW, Ww3, 50,  64,   1, 2, 1, tid, 128);
    __syncthreads();

    const int wave = tid >> 5, lane = tid & 31;
    const int hi = lane >> 4, lr = lane & 15;
    _Float16 (*chunk)[32] = schunk[wave];
    float    (*cbuf)[16]  = sC[wave];

    for (int tile = blockIdx.x * NW + wave; tile < ntiles; tile += gridDim.x * NW) {
        const int base = tile * 16;
        const int b    = base / T_SZ;          // uniform within wave (T%16==0)
        const int t0   = base - b * T_SZ;

        int ntile = tile + gridDim.x * NW;     // prefetch next tile's rows
        if (ntile < ntiles) {
            __builtin_prefetch(&intensity[(size_t)(ntile * 16 + lr) * 4], 0, 1);
            __builtin_prefetch(&weather[(size_t)(ntile * 16 + lr) * 3], 0, 1);
        }
        float pr[8];
#pragma unroll
        for (int k2 = 0; k2 < 8; ++k2) pr[k2] = pers[b * 8 + k2];

        // ---- intensity -> heads (16x8) -> fI -------------------------------
        v16h a1 = load_a1_intensity(intensity, base + lr, hi);
        v8f  c3 = run_mlp(a1, ldsW, chunk, 0, 8, 24, ba1, ba2, lr, hi, lane);
#pragma unroll
        for (int r = 0; r < 8; ++r)
            cbuf[r + 8 * hi][lr] = c3[r] + (lr < 8 ? ba3[lr] : 0.f);
        float dot = 0.f;
#pragma unroll
        for (int k2 = 0; k2 < 8; ++k2) dot += cbuf[lr][k2] * pr[k2];
        float fI = 30.f + 220.f * sigm(dot);

        // ---- weather -> wx -------------------------------------------------
        v16h a1w = load_a1_weather(weather, base + lr, hi);
        v8f  c3w = run_mlp(a1w, ldsW, chunk, 26, 34, 50, bw1, bw2, lr, hi, lane);
#pragma unroll
        for (int r = 0; r < 8; ++r)
            cbuf[r + 8 * hi][lr] = c3w[r] + (lr == 0 ? bw3[0] : 0.f);
        float wx = cbuf[lr][0];

        if (hi == 0) {
            int t = t0 + lr;
            float tg = fI * (1.f + prmB[b] * fat[t]) + prmA[b] * wx;
            target[(size_t)b * T_SZ + t] = tg;   // coalesced 16-float store
        }
    }
}

// ---------------------------------------------------------------------------
// fatigue MLP over t_norm: 1 -> 128 -> 64 -> 1  (T rows)
// ---------------------------------------------------------------------------
__global__ __launch_bounds__(256) void k_fat(
    const float* __restrict__ Wf1, const float* __restrict__ bf1,
    const float* __restrict__ Wf2, const float* __restrict__ bf2,
    const float* __restrict__ Wf3, const float* __restrict__ bf3,
    float* __restrict__ fat) {
    __shared__ float sWf2[128 * 64];
    for (int i = threadIdx.x; i < 128 * 64; i += blockDim.x) sWf2[i] = Wf2[i];
    __syncthreads();
    int t = blockIdx.x * blockDim.x + threadIdx.x;
    if (t >= T_SZ) return;
    float x = (float)t / (float)T_SZ;
    float acc[64];
#pragma unroll
    for (int k = 0; k < 64; ++k) acc[k] = bf2[k];
    for (int j = 0; j < 128; ++j) {
        float h = spf(fmaf(x, Wf1[j], bf1[j]));
        const float* wrow = &sWf2[j * 64];
#pragma unroll
        for (int k = 0; k < 64; ++k) acc[k] = fmaf(h, wrow[k], acc[k]);
    }
    float f = bf3[0];
#pragma unroll
    for (int k = 0; k < 64; ++k) f += spf(acc[k]) * Wf3[k];
    fat[t] = f;
}

// ---------------------------------------------------------------------------
// per-batch parameter decode: A,Bc,alpha,beta + pers softmax + hr0/d0
// outP layout: [A | Bc | alpha | beta | hr0 | d0], each B_SZ floats
// ---------------------------------------------------------------------------
__global__ __launch_bounds__(256) void k_params(
    const float* __restrict__ emb,
    const float* __restrict__ Wp1, const float* __restrict__ bp1,
    const float* __restrict__ Wp2, const float* __restrict__ bp2,
    const float* __restrict__ Wp3, const float* __restrict__ bp3,
    const float* __restrict__ Wpers, const float* __restrict__ bpers,
    const float* __restrict__ Wi1, const float* __restrict__ bi1,
    const float* __restrict__ Wi2, const float* __restrict__ bi2,
    float* __restrict__ outP, float* __restrict__ outPers) {
    int b = blockIdx.x * blockDim.x + threadIdx.x;
    if (b >= B_SZ) return;
    float e[16];
#pragma unroll
    for (int i = 0; i < 16; ++i) e[i] = emb[b * 16 + i];
    const float PLO[4] = {-3.f, -3.f, 0.1f, 0.1f};
    const float PHI[4] = { 5.f,  5.f, 3.0f, 3.0f};
    for (int p = 0; p < 4; ++p) {
        float h32[32];
        for (int j = 0; j < 32; ++j) {
            float s = bp1[p * 32 + j];
#pragma unroll
            for (int i = 0; i < 16; ++i) s += e[i] * Wp1[p * 512 + i * 32 + j];
            h32[j] = spf(s);
        }
        float h8[8];
        for (int k = 0; k < 8; ++k) {
            float s = bp2[p * 8 + k];
            for (int j = 0; j < 32; ++j) s += h32[j] * Wp2[p * 256 + j * 8 + k];
            h8[k] = spf(s);
        }
        float raw = bp3[p];
#pragma unroll
        for (int k = 0; k < 8; ++k) raw += h8[k] * Wp3[p * 8 + k];
        outP[p * B_SZ + b] = PLO[p] + (PHI[p] - PLO[p]) * sigm(raw);
    }
    // pers softmax
    float lg[8];
    for (int k = 0; k < 8; ++k) {
        float s = bpers[k];
#pragma unroll
        for (int i = 0; i < 16; ++i) s += e[i] * Wpers[i * 8 + k];
        lg[k] = s;
    }
    float m = lg[0];
#pragma unroll
    for (int k = 1; k < 8; ++k) m = fmaxf(m, lg[k]);
    float sum = 0.f;
#pragma unroll
    for (int k = 0; k < 8; ++k) { lg[k] = __expf(lg[k] - m); sum += lg[k]; }
#pragma unroll
    for (int k = 0; k < 8; ++k) outPers[b * 8 + k] = lg[k] / sum;
    // initial state
    float hd0 = bi2[0], hd1 = bi2[1];
    for (int j = 0; j < 32; ++j) {
        float s = bi1[j];
#pragma unroll
        for (int i = 0; i < 16; ++i) s += e[i] * Wi1[i * 32 + j];
        float h = spf(s);
        hd0 += h * Wi2[j * 2 + 0];
        hd1 += h * Wi2[j * 2 + 1];
    }
    outP[4 * B_SZ + b] = 30.f + 220.f * sigm(hd0);
    outP[5 * B_SZ + b] = 30.f + 220.f * sigm(hd1);
}

// ---------------------------------------------------------------------------
// Chunked parallel scan of  d' = (1-b)d + b*tg ;  hr' = (1-a)hr + a*d'
// state s=(hr,d): s' = M s + u,  M=[[1-a, a(1-b)],[0,1-b]], u=(a*b*tg, b*tg)
// ---------------------------------------------------------------------------
__global__ __launch_bounds__(256) void k_scan1(
    const float* __restrict__ al_, const float* __restrict__ be_,
    const float* __restrict__ target,
    float* __restrict__ cp00, float* __restrict__ cp01, float* __restrict__ cp11,
    float* __restrict__ cw0,  float* __restrict__ cw1) {
    int idx = blockIdx.x * blockDim.x + threadIdx.x;
    if (idx >= B_SZ * NC) return;
    int b = idx / NC, c = idx % NC;
    float al = al_[b], be = be_[b];
    float m00 = 1.f - al, m01 = al * (1.f - be), m11 = 1.f - be;
    float p00 = 1.f, p01 = 0.f, p11 = 1.f, w0 = 0.f, w1 = 0.f;
    const float* tg = target + (size_t)b * T_SZ + c * CL;
    for (int j = 0; j < CL; ++j) {
        float u   = tg[j];
        float nw0 = fmaf(m00, w0, fmaf(m01, w1, al * be * u));
        float nw1 = fmaf(m11, w1, be * u);
        w0 = nw0; w1 = nw1;
        p01 = fmaf(m00, p01, m01 * p11);
        p00 *= m00; p11 *= m11;
    }
    cp00[idx] = p00; cp01[idx] = p01; cp11[idx] = p11;
    cw0[idx] = w0;   cw1[idx] = w1;
}

__global__ __launch_bounds__(256) void k_scanB(
    const float* __restrict__ hr0, const float* __restrict__ d0,
    const float* __restrict__ cp00, const float* __restrict__ cp01,
    const float* __restrict__ cp11, const float* __restrict__ cw0,
    const float* __restrict__ cw1,
    float* __restrict__ sh0, float* __restrict__ sh1) {
    int b = blockIdx.x * blockDim.x + threadIdx.x;
    if (b >= B_SZ) return;
    float hr = hr0[b], d = d0[b];
    for (int c = 0; c < NC; ++c) {
        int i = b * NC + c;
        sh0[i] = hr; sh1[i] = d;
        float nhr = fmaf(cp00[i], hr, fmaf(cp01[i], d, cw0[i]));
        float nd  = fmaf(cp11[i], d, cw1[i]);
        hr = nhr; d = nd;
    }
}

__global__ __launch_bounds__(256) void k_scan2(
    const float* __restrict__ al_, const float* __restrict__ be_,
    const float* __restrict__ target,
    const float* __restrict__ sh0, const float* __restrict__ sh1,
    float* __restrict__ out) {
    int idx = blockIdx.x * blockDim.x + threadIdx.x;
    if (idx >= B_SZ * NC) return;
    int b = idx / NC, c = idx % NC;
    float al = al_[b], be = be_[b];
    float hr = sh0[idx], d = sh1[idx];
    const float* tg = target + (size_t)b * T_SZ + c * CL;
    float* o = out + (size_t)b * T_SZ + c * CL;
    for (int j = 0; j < CL; ++j) {
        float u = tg[j];
        d  = fmaf(be, u - d, d);       // d += beta*(target - d)
        hr = fmaf(al, d - hr, hr);     // hr += alpha*(d - hr)
        o[j] = hr;
    }
}

// ---------------------------------------------------------------------------
extern "C" void kernel_launch(void* const* d_in, const int* in_sizes, int n_in,
                              void* d_out, int out_size, void* d_ws, size_t ws_size,
                              hipStream_t stream) {
    const float* intensity = (const float*)d_in[0];
    const float* weather   = (const float*)d_in[1];
    const float* emb       = (const float*)d_in[2];
    const float* Wp1 = (const float*)d_in[3],  *bp1 = (const float*)d_in[4];
    const float* Wp2 = (const float*)d_in[5],  *bp2 = (const float*)d_in[6];
    const float* Wp3 = (const float*)d_in[7],  *bp3 = (const float*)d_in[8];
    const float* Wa1 = (const float*)d_in[9],  *ba1 = (const float*)d_in[10];
    const float* Wa2 = (const float*)d_in[11], *ba2 = (const float*)d_in[12];
    const float* Wa3 = (const float*)d_in[13], *ba3 = (const float*)d_in[14];
    const float* Wpers = (const float*)d_in[15], *bpers = (const float*)d_in[16];
    const float* Ww1 = (const float*)d_in[17], *bw1 = (const float*)d_in[18];
    const float* Ww2 = (const float*)d_in[19], *bw2 = (const float*)d_in[20];
    const float* Ww3 = (const float*)d_in[21], *bw3 = (const float*)d_in[22];
    const float* Wf1 = (const float*)d_in[23], *bf1 = (const float*)d_in[24];
    const float* Wf2 = (const float*)d_in[25], *bf2 = (const float*)d_in[26];
    const float* Wf3 = (const float*)d_in[27], *bf3 = (const float*)d_in[28];
    const float* Wi1 = (const float*)d_in[29], *bi1 = (const float*)d_in[30];
    const float* Wi2 = (const float*)d_in[31], *bi2 = (const float*)d_in[32];
    float* out = (float*)d_out;

    // workspace layout (floats); total ~7.93 MB
    float* ws     = (float*)d_ws;
    float* wFat   = ws;                           // T_SZ
    float* wPar   = wFat + T_SZ;                  // 6*B_SZ : A,Bc,alpha,beta,hr0,d0
    float* wPers  = wPar + 6 * B_SZ;              // B_SZ*8
    float* wTgt   = wPers + B_SZ * 8;             // B_SZ*T_SZ
    float* wP00   = wTgt + (size_t)B_SZ * T_SZ;   // B_SZ*NC each
    float* wP01   = wP00 + B_SZ * NC;
    float* wP11   = wP01 + B_SZ * NC;
    float* wW0    = wP11 + B_SZ * NC;
    float* wW1    = wW0 + B_SZ * NC;
    float* wSh0   = wW1 + B_SZ * NC;              // B_SZ*NC
    float* wSh1   = wSh0 + B_SZ * NC;

    k_fat<<<(T_SZ + 255) / 256, 256, 0, stream>>>(Wf1, bf1, Wf2, bf2, Wf3, bf3, wFat);
    k_params<<<(B_SZ + 255) / 256, 256, 0, stream>>>(
        emb, Wp1, bp1, Wp2, bp2, Wp3, bp3, Wpers, bpers, Wi1, bi1, Wi2, bi2,
        wPar, wPers);
    k_mlp<<<2048, 128, 0, stream>>>(
        intensity, weather,
        Wa1, ba1, Wa2, ba2, Wa3, ba3,
        Ww1, bw1, Ww2, bw2, Ww3, bw3,
        wPers, wFat, /*A*/ wPar, /*Bc*/ wPar + B_SZ,
        wTgt, NTILES);
    k_scan1<<<(B_SZ * NC + 255) / 256, 256, 0, stream>>>(
        wPar + 2 * B_SZ, wPar + 3 * B_SZ, wTgt, wP00, wP01, wP11, wW0, wW1);
    k_scanB<<<(B_SZ + 255) / 256, 256, 0, stream>>>(
        wPar + 4 * B_SZ, wPar + 5 * B_SZ, wP00, wP01, wP11, wW0, wW1, wSh0, wSh1);
    k_scan2<<<(B_SZ * NC + 255) / 256, 256, 0, stream>>>(
        wPar + 2 * B_SZ, wPar + 3 * B_SZ, wTgt, wSh0, wSh1, out);
}
